// Gemma3nAudioAttention_2783138808551
// MI455X (gfx1250) — matrix operand order repaired and checked by reading, not compiled, not run on hardware
//
#include <hip/hip_runtime.h>
#include <math.h>

// ---------- types ----------
typedef __attribute__((ext_vector_type(16))) __bf16   v16bf;
typedef __attribute__((ext_vector_type(8)))  float    v8f;
typedef __attribute__((ext_vector_type(4)))  unsigned int v4u;
typedef __attribute__((ext_vector_type(4)))  float    v4f;
typedef __attribute__((ext_vector_type(2)))  unsigned int v2u;

union Frag { v16bf bf; v4u q[2]; unsigned int u[8]; };
union Acc  { v8f v; float f[8]; };

#define T_LEN   8192
#define BATCH   8
#define HID     1536
#define NHEADS  8
#define HDIM    192
#define QKVN    4608     // 3*HID
#define UBLK    683      // ceil(8192/12)
#define GM      65536    // BATCH*T_LEN

__device__ __forceinline__ unsigned short f2bf(float x) {
  union { float f; unsigned int u; } a; a.f = x;
  unsigned int r = a.u + 0x7FFFu + ((a.u >> 16) & 1u);
  return (unsigned short)(r >> 16);
}

// ---------- kernel 1: fp32 -> bf16 convert of hidden states ----------
__global__ void cvt_kernel(const float* __restrict__ in,
                           unsigned short* __restrict__ out, long n4) {
  long i = (long)blockIdx.x * blockDim.x + threadIdx.x;
  if (i >= n4) return;
  v4f v = ((const v4f*)in)[i];
  v2u o;
  o[0] = (unsigned int)f2bf(v[0]) | ((unsigned int)f2bf(v[1]) << 16);
  o[1] = (unsigned int)f2bf(v[2]) | ((unsigned int)f2bf(v[3]) << 16);
  ((v2u*)out)[i] = o;
}

// ---------- kernel 2: pack W_q|W_k|W_v transposed -> Wt[n][k] bf16 ----------
__global__ void packw_kernel(const float* __restrict__ wq,
                             const float* __restrict__ wk,
                             const float* __restrict__ wv,
                             unsigned short* __restrict__ Wt) {
  long i = (long)blockIdx.x * blockDim.x + threadIdx.x;   // over QKVN*HID
  if (i >= (long)QKVN * HID) return;
  int n = (int)(i / HID), k = (int)(i % HID);
  int nn;
  const float* w;
  if (n < HID)            { w = wq; nn = n; }
  else if (n < 2 * HID)   { w = wk; nn = n - HID; }
  else                    { w = wv; nn = n - 2 * HID; }
  Wt[i] = f2bf(w[(size_t)k * HID + nn]);
}

// ---------- kernel 3: q per-dim scale = Q_SCALE * softplus(pds) ----------
__global__ void qscale_kernel(const float* __restrict__ pds, float* __restrict__ qs) {
  int h = threadIdx.x;
  if (h < HDIM) {
    float x = pds[h];
    float sp = (x > 20.f) ? x : log1pf(__expf(x));
    qs[h] = (0.07216878364870322f / 0.69314718055994531f) * sp;  // HDIM^-0.5 / ln2
  }
}

// ---------- kernel 4: sin/cos positional embedding @ w_pos -> bf16 [16][1536] ----------
__global__ void semb_kernel(const float* __restrict__ wpos,
                            unsigned short* __restrict__ semb) {
  int nh = blockIdx.x * blockDim.x + threadIdx.x;   // 0..1535
  int f  = blockIdx.y;                              // 0..15 (13..15 zero pad)
  if (nh >= HID) return;
  float acc = 0.f;
  if (f < 13) {
    float pos = -(float)f;                          // pos = arange(0,-13,-1)
    const float log_inc = 9.210340371976184f / 767.0f;
    for (int d = 0; d < 768; ++d) {
      float inv = __expf(-log_inc * (float)d);
      float sc  = pos * inv;
      acc += sinf(sc) * wpos[(size_t)d * HID + nh]
           + cosf(sc) * wpos[(size_t)(d + 768) * HID + nh];
    }
  }
  semb[(size_t)f * HID + nh] = f2bf(acc);
}

// ---------- kernel 5: fused QKV GEMM, bf16 WMMA ----------
// X: [GM][HID] bf16, Wt: [QKVN][HID] bf16 (transposed weights), out QKV: [GM][QKVN] bf16
__global__ __launch_bounds__(128) void gemm_qkv(const unsigned short* __restrict__ X,
                                                const unsigned short* __restrict__ Wt,
                                                const float* __restrict__ qs,
                                                unsigned short* __restrict__ QKV) {
  const int wave = threadIdx.x >> 5, lane = threadIdx.x & 31;
  const int lh = lane >> 4, lm = lane & 15;
  const int wm = wave & 1, wn = wave >> 1;
  const int m0 = blockIdx.x * 64 + wm * 32;
  const int n0 = blockIdx.y * 128 + wn * 64;
  const int kb = lh * 8, kbB = lh * 16;

  Acc acc[8];
  #pragma unroll
  for (int i = 0; i < 8; ++i)
    #pragma unroll
    for (int j = 0; j < 8; ++j) acc[i].f[j] = 0.f;

  const unsigned short* arow0 = X + (size_t)(m0 + lm) * HID;
  const unsigned short* arow1 = X + (size_t)(m0 + 16 + lm) * HID;

  for (int kk = 0; kk < HID; kk += 32) {
    Frag a0, a1;
    a0.q[0] = *(const v4u*)(arow0 + kk + kbB);
    a0.q[1] = *(const v4u*)(arow0 + kk + 8 + kbB);
    a1.q[0] = *(const v4u*)(arow1 + kk + kbB);
    a1.q[1] = *(const v4u*)(arow1 + kk + 8 + kbB);
    #pragma unroll
    for (int nt = 0; nt < 4; ++nt) {
      Frag b;
      const unsigned short* brow = Wt + (size_t)(n0 + nt * 16 + lm) * HID + kk + kbB;
      b.q[0] = *(const v4u*)(brow);
      b.q[1] = *(const v4u*)(brow + 8);
      acc[nt * 2 + 0].v = __builtin_amdgcn_wmma_f32_16x16x32_bf16(
          false, a0.bf, false, b.bf, (short)0, acc[nt * 2 + 0].v, false, false);
      acc[nt * 2 + 1].v = __builtin_amdgcn_wmma_f32_16x16x32_bf16(
          false, a1.bf, false, b.bf, (short)0, acc[nt * 2 + 1].v, false, false);
    }
  }

  #pragma unroll
  for (int nt = 0; nt < 4; ++nt) {
    int n = n0 + nt * 16 + lm;
    float scale = (n < HID) ? qs[n % HDIM] : 1.0f;   // scale applies to Q columns only
    #pragma unroll
    for (int mi = 0; mi < 2; ++mi) {
      int mbase = m0 + mi * 16 + 8 * lh;
      #pragma unroll
      for (int r = 0; r < 8; ++r)
        QKV[(size_t)(mbase + r) * QKVN + n] = f2bf(acc[nt * 2 + mi].f[r] * scale);
    }
  }
}

// ---------- kernel 6: chunked local attention ----------
// one block = one (b,u) chunk; wave w = head w.
__global__ __launch_bounds__(256) void attn_kernel(const unsigned short* __restrict__ QKV,
                                                   const unsigned short* __restrict__ semb,
                                                   const unsigned char* __restrict__ maskv,
                                                   float* __restrict__ out) {
  __shared__ __align__(16) unsigned short pstage[NHEADS][512];  // [w][c] bf16, 16x32
  const int u = blockIdx.x, b = blockIdx.y;
  const int head = threadIdx.x >> 5, lane = threadIdx.x & 31;
  const int lh = lane >> 4, lm = lane & 15;
  const int kb = lh * 8, kbB = lh * 16;

  // ---- Q A-fragment row (m = lm) ----
  const int tq = u * 12 + lm;
  const bool qv = (lm < 12) && (tq < T_LEN);
  const unsigned short* qbase =
      QKV + ((size_t)(b * T_LEN + (qv ? tq : 0))) * QKVN + head * HDIM;

  // ---- K context validity + base ptrs (n = c = ct*16+lm) ----
  bool kval[2];
  const unsigned short* kbase[2];
  #pragma unroll
  for (int ct = 0; ct < 2; ++ct) {
    int c = ct * 16 + lm;
    int t = u * 12 - 12 + c;
    bool ok = (c < 24) && (t >= 0) && (t < T_LEN);
    if (ok) ok = (maskv[(size_t)b * T_LEN + t] == 0);
    kval[ct] = ok;
    kbase[ct] = QKV + ((size_t)(b * T_LEN + (ok ? t : 0))) * QKVN + HID + head * HDIM;
  }
  const unsigned short* sbase = semb + (size_t)lm * HID + head * HDIM;

  Acc ac0, ac1, bd;
  #pragma unroll
  for (int j = 0; j < 8; ++j) { ac0.f[j] = 0.f; ac1.f[j] = 0.f; bd.f[j] = 0.f; }

  #pragma unroll
  for (int h0 = 0; h0 < HDIM; h0 += 32) {
    Frag a;
    if (qv) { a.q[0] = *(const v4u*)(qbase + h0 + kbB);
              a.q[1] = *(const v4u*)(qbase + h0 + 8 + kbB); }
    else    { a.q[0] = (v4u)0u; a.q[1] = (v4u)0u; }
    Frag s;
    s.q[0] = *(const v4u*)(sbase + h0 + kbB);
    s.q[1] = *(const v4u*)(sbase + h0 + kbB + 8);
    bd.v = __builtin_amdgcn_wmma_f32_16x16x32_bf16(false, a.bf, false, s.bf,
                                                   (short)0, bd.v, false, false);
    Frag k0, k1;
    if (kval[0]) { k0.q[0] = *(const v4u*)(kbase[0] + h0 + kbB);
                   k0.q[1] = *(const v4u*)(kbase[0] + h0 + kbB + 8); }
    else         { k0.q[0] = (v4u)0u; k0.q[1] = (v4u)0u; }
    if (kval[1]) { k1.q[0] = *(const v4u*)(kbase[1] + h0 + kbB);
                   k1.q[1] = *(const v4u*)(kbase[1] + h0 + kbB + 8); }
    else         { k1.q[0] = (v4u)0u; k1.q[1] = (v4u)0u; }
    ac0.v = __builtin_amdgcn_wmma_f32_16x16x32_bf16(false, a.bf, false, k0.bf,
                                                    (short)0, ac0.v, false, false);
    ac1.v = __builtin_amdgcn_wmma_f32_16x16x32_bf16(false, a.bf, false, k1.bf,
                                                    (short)0, ac1.v, false, false);
  }

  // ---- logits: +bd (relative shift == select f=c-w in [0,12]), softcap, mask, softmax ----
  #pragma unroll
  for (int r = 0; r < 8; ++r) {
    int w  = r + 8 * lh;
    int c0 = lm, c1 = 16 + lm;
    int f0 = c0 - w, f1 = c1 - w;
    float b0 = __shfl(bd.f[r], (f0 & 15) + 16 * lh, 32);
    float b1 = __shfl(bd.f[r], (f1 & 15) + 16 * lh, 32);
    if (f0 < 0 || f0 > 12) b0 = 0.f;
    if (f1 < 0 || f1 > 12) b1 = 0.f;
    float l0 = tanhf((ac0.f[r] + b0) * 0.02f) * 50.f;
    float l1 = tanhf((ac1.f[r] + b1) * 0.02f) * 50.f;
    bool wv = (w < 12) && (u * 12 + w < T_LEN);
    bool v0 = wv && kval[0] && (w <= c0) && (c0 <= w + 12);
    bool v1 = wv && kval[1] && (w <= c1) && (c1 <= w + 12);
    l0 = v0 ? l0 : -3.0e38f;
    l1 = v1 ? l1 : -3.0e38f;
    float mx = fmaxf(l0, l1);
    #pragma unroll
    for (int sft = 1; sft < 16; sft <<= 1) mx = fmaxf(mx, __shfl_xor(mx, sft, 32));
    float e0 = __expf(l0 - mx), e1 = __expf(l1 - mx);
    float sm = e0 + e1;
    #pragma unroll
    for (int sft = 1; sft < 16; sft <<= 1) sm += __shfl_xor(sm, sft, 32);
    float inv = 1.0f / sm;
    pstage[head][w * 32 + c0] = f2bf(e0 * inv);
    pstage[head][w * 32 + c1] = f2bf(e1 * inv);
  }
  __syncthreads();

  // ---- PV: A = probs (from LDS), B = V (gathered u16 pairs) ----
  Frag pa;
  const unsigned short* pp = &pstage[head][lm * 32 + kbB];
  pa.q[0] = *(const v4u*)pp;
  pa.q[1] = *(const v4u*)(pp + 8);

  unsigned vb16 = 0;
  #pragma unroll
  for (int jj = 0; jj < 16; ++jj) {
    int c = kbB + jj;
    int t = u * 12 - 12 + c;
    bool ok = (c < 24) && (t >= 0) && (t < T_LEN);
    if (ok) ok = (maskv[(size_t)b * T_LEN + t] == 0);
    vb16 |= ok ? (1u << jj) : 0u;
  }
  const int tc0 = u * 12 - 12 + kbB;

  #pragma unroll
  for (int ht = 0; ht < 12; ++ht) {
    int h = ht * 16 + lm;
    Frag vf;
    #pragma unroll
    for (int j = 0; j < 8; ++j) {
      int j0 = 2 * j;
      unsigned lo = 0, hi = 0;
      if ((vb16 >> j0) & 1u)
        lo = QKV[((size_t)(b * T_LEN + tc0 + j0)) * QKVN + 2 * HID + head * HDIM + h];
      if ((vb16 >> (j0 + 1)) & 1u)
        hi = QKV[((size_t)(b * T_LEN + tc0 + j0 + 1)) * QKVN + 2 * HID + head * HDIM + h];
      vf.u[j] = lo | (hi << 16);
    }
    Acc z;
    #pragma unroll
    for (int j = 0; j < 8; ++j) z.f[j] = 0.f;
    z.v = __builtin_amdgcn_wmma_f32_16x16x32_bf16(false, pa.bf, false, vf.bf,
                                                  (short)0, z.v, false, false);
    #pragma unroll
    for (int r = 0; r < 8; ++r) {
      int w = r + 8 * lh;
      int t = u * 12 + w;
      if (w < 12 && t < T_LEN)
        out[(((size_t)b * T_LEN + t) * NHEADS + head) * HDIM + h] = z.f[r];
    }
  }
}

// ---------- launcher ----------
extern "C" void kernel_launch(void* const* d_in, const int* in_sizes, int n_in,
                              void* d_out, int out_size, void* d_ws, size_t ws_size,
                              hipStream_t stream) {
  (void)in_sizes; (void)n_in; (void)out_size;
  const float*         hs    = (const float*)d_in[0];
  const unsigned char* maskv = (const unsigned char*)d_in[1];
  const float*         wq    = (const float*)d_in[2];
  const float*         wk    = (const float*)d_in[3];
  const float*         wv    = (const float*)d_in[4];
  const float*         wpos  = (const float*)d_in[5];
  const float*         pds   = (const float*)d_in[6];
  float*               out   = (float*)d_out;

  char* ws = (char*)d_ws;
  size_t off = 0;
  auto alloc = [&](size_t bytes) -> char* {
    char* p = ws + off;
    off += (bytes + 255) & ~(size_t)255;
    return p;
  };
  unsigned short* Xb  = (unsigned short*)alloc((size_t)GM * HID * 2);
  unsigned short* Wt  = (unsigned short*)alloc((size_t)QKVN * HID * 2);
  unsigned short* QKV = (unsigned short*)alloc((size_t)GM * QKVN * 2);
  float*          qs  = (float*)alloc(256 * sizeof(float));
  unsigned short* sem = (unsigned short*)alloc((size_t)16 * HID * 2);
  if (off > ws_size) return;  // workspace too small; bail deterministically

  long n4 = (long)GM * HID / 4;
  cvt_kernel<<<(unsigned)(n4 / 256), 256, 0, stream>>>(hs, Xb, n4);
  packw_kernel<<<(unsigned)((long)QKVN * HID / 256), 256, 0, stream>>>(wq, wk, wv, Wt);
  qscale_kernel<<<1, 192, 0, stream>>>(pds, qs);
  semb_kernel<<<dim3(HID / 256, 16), 256, 0, stream>>>(wpos, sem);
  gemm_qkv<<<dim3(GM / 64, QKVN / 128), 128, 0, stream>>>(Xb, Wt, qs, QKV);
  attn_kernel<<<dim3(UBLK, BATCH), 256, 0, stream>>>(QKV, sem, maskv, out);
}
